// GNN_PyG_15101105013187
// MI455X (gfx1250) — matrix operand order repaired and checked by reading, not compiled, not run on hardware
//
#include <hip/hip_runtime.h>
#include <hip/hip_bf16.h>

// ---------------------------------------------------------------------------
// GCN inference for MI455X (gfx1250): WMMA bf16 GEMMs + L2-resident edge phase
// ---------------------------------------------------------------------------

typedef __attribute__((ext_vector_type(16))) __bf16         v16bf;
typedef __attribute__((ext_vector_type(16))) unsigned short v16u;
typedef __attribute__((ext_vector_type(8)))  float          v8f;

__device__ __forceinline__ unsigned short f32_to_bf16_rne(float f) {
    unsigned u = __builtin_bit_cast(unsigned, f);
    unsigned r = u + 0x7FFFu + ((u >> 16) & 1u);   // round-to-nearest-even
    return (unsigned short)(r >> 16);
}

// ---------------------------------------------------------------------------
// Tiled GEMM: out[nrows x NC] = A[nrows x K] * W[K x NC] (+bias, +relu)
// f32 in memory, bf16 WMMA compute with f32 accumulation.
// 256 threads = 8 waves; each wave computes 16 rows x NC cols.
//
// LDS holds W pre-swizzled into WMMA B-fragment order:
//   sW[((kk*NT + t)*32 + lane)*16 + j]  (16 contiguous bf16 per lane-fragment)
// so each B fragment is one aligned 32-byte LDS load (2x ds_load_b128).
//
// A rows are CLAMPED (not masked): A row M only feeds D row M, and
// out-of-range rows are never stored, so padding lanes may read a duplicate
// row. This keeps every A access an unconditional global_load_b128.
// ---------------------------------------------------------------------------
template<int K, int NC, bool BIAS, bool RELU>
__global__ __launch_bounds__(256)
void gemm_bf16_wmma(const float* __restrict__ A, const float* __restrict__ W,
                    const float* __restrict__ bias, float* __restrict__ out,
                    int nrows)
{
    constexpr int NT   = (NC + 15) / 16;      // 16-col tiles per wave
    constexpr int KT   = K / 32;              // K-steps of 32
    constexpr int FRAG = KT * NT * 32 * 16;   // total staged bf16 elements

    __shared__ unsigned short sW[FRAG];       // bf16 bits, fragment-ordered

    const int tid = threadIdx.x;

    // ---- cooperative stage: f32 -> bf16, swizzled to B-fragment layout ----
    // destination pair index i -> (tile, lane, j-pair); B fragment element j of
    // lane (half,l16) in tile (kk,t) is W[kk*32 + (j/8)*16 + half*8 + j%8][t*16 + l16].
    {
        unsigned* sW32 = (unsigned*)sW;
        for (int i = tid; i < FRAG / 2; i += 256) {
            const int j2   = i & 7;                 // bf16 pair within fragment
            const int ln   = (i >> 3) & 31;         // lane
            const int tile = i >> 8;                // kk*NT + t
            const int t    = tile % NT;
            const int kk   = tile / NT;
            const int k0   = kk * 32 + (j2 >> 2) * 16 + (ln >> 4) * 8 + (j2 & 3) * 2;
            const int c    = t * 16 + (ln & 15);
            float va = (c < NC) ? W[(size_t)k0 * NC + c]       : 0.0f;
            float vb = (c < NC) ? W[(size_t)(k0 + 1) * NC + c] : 0.0f;
            sW32[i] = (unsigned)f32_to_bf16_rne(va) |
                      ((unsigned)f32_to_bf16_rne(vb) << 16);
        }
    }
    __syncthreads();

    const int wave = tid >> 5;
    const int lane = tid & 31;
    const int half = lane >> 4;     // 0: K-low half, 1: K-high half
    const int l16  = lane & 15;

    const int row0 = (blockIdx.x * 8 + wave) * 16;
    int arow = row0 + l16;                      // A row this lane feeds
    if (arow > nrows - 1) arow = nrows - 1;     // clamp: no masked loads
    const float* aptr = A + (size_t)arow * K;

    const v16u* __restrict__ sWB = (const v16u*)sW;   // 32B fragments

    v8f acc[NT] = {};

    #pragma unroll
    for (int kk = 0; kk < KT; ++kk) {
        // ---- A fragment (16x32 bf16, ISA lane layout), float4 loads ----
        const float4* ap = (const float4*)(aptr + kk * 32 + half * 8);
        const float4 p0 = ap[0];   // K = base+0..3
        const float4 p1 = ap[1];   // K = base+4..7
        const float4 p2 = ap[4];   // K = base+16..19
        const float4 p3 = ap[5];   // K = base+20..23

        v16u au;
        au[0]  = f32_to_bf16_rne(p0.x); au[1]  = f32_to_bf16_rne(p0.y);
        au[2]  = f32_to_bf16_rne(p0.z); au[3]  = f32_to_bf16_rne(p0.w);
        au[4]  = f32_to_bf16_rne(p1.x); au[5]  = f32_to_bf16_rne(p1.y);
        au[6]  = f32_to_bf16_rne(p1.z); au[7]  = f32_to_bf16_rne(p1.w);
        au[8]  = f32_to_bf16_rne(p2.x); au[9]  = f32_to_bf16_rne(p2.y);
        au[10] = f32_to_bf16_rne(p2.z); au[11] = f32_to_bf16_rne(p2.w);
        au[12] = f32_to_bf16_rne(p3.x); au[13] = f32_to_bf16_rne(p3.y);
        au[14] = f32_to_bf16_rne(p3.z); au[15] = f32_to_bf16_rne(p3.w);
        v16bf afrag = __builtin_bit_cast(v16bf, au);

        #pragma unroll
        for (int t = 0; t < NT; ++t) {
            // ---- B fragment: one aligned 32-byte LDS load ----
            v16u  bu    = sWB[(kk * NT + t) * 32 + lane];
            v16bf bfrag = __builtin_bit_cast(v16bf, bu);

            acc[t] = __builtin_amdgcn_wmma_f32_16x16x32_bf16(
                         false, afrag, false, bfrag,
                         (short)0, acc[t], false, false);
        }
    }

    // ---- epilogue: D tile store (VGPR r -> row half*8+r, col lane&15) ----
    if (row0 + 16 <= nrows) {
        // fast path (wave-uniform): no row guards
        #pragma unroll
        for (int t = 0; t < NT; ++t) {
            const int col = t * 16 + l16;
            if (col < NC) {
                const float b = BIAS ? bias[col] : 0.0f;
                #pragma unroll
                for (int r = 0; r < 8; ++r) {
                    const int row = row0 + half * 8 + r;
                    float v = acc[t][r] + b;
                    if (RELU) v = fmaxf(v, 0.0f);
                    out[(size_t)row * NC + col] = v;
                }
            }
        }
    } else {
        // tail block: per-row guard
        #pragma unroll
        for (int t = 0; t < NT; ++t) {
            const int col = t * 16 + l16;
            if (col < NC) {
                const float b = BIAS ? bias[col] : 0.0f;
                #pragma unroll
                for (int r = 0; r < 8; ++r) {
                    const int row = row0 + half * 8 + r;
                    if (row < nrows) {
                        float v = acc[t][r] + b;
                        if (RELU) v = fmaxf(v, 0.0f);
                        out[(size_t)row * NC + col] = v;
                    }
                }
            }
        }
    }
}

// ---------------------------------------------------------------------------
// Graph / elementwise kernels
// ---------------------------------------------------------------------------
__global__ void zero_f32(float* __restrict__ p, int n) {
    int i = blockIdx.x * blockDim.x + threadIdx.x;
    if (i < n) p[i] = 0.0f;
}

__global__ void deg_kernel(const int* __restrict__ dst, float* __restrict__ deg, int E) {
    int i = blockIdx.x * blockDim.x + threadIdx.x;
    if (i < E) atomicAdd(&deg[dst[i]], 1.0f);
}

__global__ void dinv_kernel(float* __restrict__ d, int n) {
    int i = blockIdx.x * blockDim.x + threadIdx.x;
    if (i < n) {
        float v = d[i];
        d[i] = (v > 0.0f) ? rsqrtf(fmaxf(v, 1.0f)) : 0.0f;
    }
}

__global__ void norm_kernel(const int* __restrict__ src, const int* __restrict__ dst,
                            const float* __restrict__ dinv, float* __restrict__ norm, int E) {
    int i = blockIdx.x * blockDim.x + threadIdx.x;
    if (i < E) norm[i] = dinv[src[i]] * dinv[dst[i]];
}

// acc[n][f] = bias[f]   (H == 64)
__global__ void init_acc(float* __restrict__ acc, const float* __restrict__ bias, int n) {
    int i = blockIdx.x * blockDim.x + threadIdx.x;
    if (i < n) acc[i] = bias[i & 63];
}

// 16 lanes per edge, float4 gather + 4 f32 atomics (coalesced 256B per edge).
// hw (25.6MB) and acc fit in the 192MB L2 -> random access stays on-chip.
__global__ void scatter_kernel(const float* __restrict__ hw, const int* __restrict__ src,
                               const int* __restrict__ dst, const float* __restrict__ norm,
                               float* __restrict__ acc, int E)
{
    size_t idx = (size_t)blockIdx.x * blockDim.x + threadIdx.x;
    if (idx >= (size_t)E * 16) return;
    const int e = (int)(idx >> 4);
    const int q = (int)(idx & 15);
    const int s = src[e], d = dst[e];
    const float w = norm[e];
    const float4 v = *(const float4*)(hw + (size_t)s * 64 + q * 4);
    float* dp = acc + (size_t)d * 64 + q * 4;
    atomicAdd(dp + 0, v.x * w);
    atomicAdd(dp + 1, v.y * w);
    atomicAdd(dp + 2, v.z * w);
    atomicAdd(dp + 3, v.w * w);
}

// residual + running-stat BatchNorm + ReLU, updates h_last in place (H == 64)
__global__ void post_kernel(const float* __restrict__ acc, float* __restrict__ hlast,
                            float* __restrict__ h,
                            const float* __restrict__ g, const float* __restrict__ b,
                            const float* __restrict__ m, const float* __restrict__ v,
                            int n, int use_res)
{
    int i = blockIdx.x * blockDim.x + threadIdx.x;
    if (i >= n) return;
    const int f = i & 63;
    float x = acc[i];
    if (use_res) x += hlast[i];
    hlast[i] = x;
    float y = g[f] * (x - m[f]) * rsqrtf(v[f] + 1e-5f) + b[f];
    h[i] = fmaxf(y, 0.0f);
}

// ---------------------------------------------------------------------------
extern "C" void kernel_launch(void* const* d_in, const int* in_sizes, int n_in,
                              void* d_out, int out_size, void* d_ws, size_t ws_size,
                              hipStream_t stream)
{
    (void)n_in; (void)out_size; (void)ws_size;

    const float* x      = (const float*)d_in[0];
    const int*   edge   = (const int*)  d_in[1];
    const float* enc_W  = (const float*)d_in[2];
    const float* enc_b  = (const float*)d_in[3];
    const float* conv_W = (const float*)d_in[4];   // [3,64,64]
    const float* conv_b = (const float*)d_in[5];   // [3,64]
    const float* bn_g   = (const float*)d_in[6];
    const float* bn_b   = (const float*)d_in[7];
    const float* bn_m   = (const float*)d_in[8];
    const float* bn_v   = (const float*)d_in[9];
    const float* pred_W = (const float*)d_in[10];
    const float* pred_b = (const float*)d_in[11];
    float* out = (float*)d_out;

    const int FIN = 128, H = 64, L = 3;
    const int N = in_sizes[0] / FIN;
    const int E = in_sizes[1] / 2;
    const size_t NH = (size_t)N * H;

    // workspace layout (f32)
    float* ws    = (float*)d_ws;
    float* dinv  = ws;                 // N       (doubles as degree buffer)
    float* norm  = dinv + N;           // E
    float* h     = norm + E;           // N*H
    float* hw    = h    + NH;          // N*H
    float* acc   = hw   + NH;          // N*H
    float* hlast = acc  + NH;          // N*H

    const int* src = edge;
    const int* dst = edge + E;

    // --- GCN normalization: deg -> deg^{-1/2} -> per-edge norm ---
    zero_f32 <<<(N + 255) / 256, 256, 0, stream>>>(dinv, N);
    deg_kernel<<<(E + 255) / 256, 256, 0, stream>>>(dst, dinv, E);
    dinv_kernel<<<(N + 255) / 256, 256, 0, stream>>>(dinv, N);
    norm_kernel<<<(E + 255) / 256, 256, 0, stream>>>(src, dst, dinv, norm, E);

    // --- encoder: h = relu(x @ enc_W + enc_b) ---
    gemm_bf16_wmma<128, 64, true, true>
        <<<(N + 127) / 128, 256, 0, stream>>>(x, enc_W, enc_b, h, N);

    const int nhBlocks = (int)((NH + 255) / 256);
    const int scBlocks = (int)(((size_t)E * 16 + 255) / 256);

    for (int l = 0; l < L; ++l) {
        // hw = h @ conv_W[l]
        gemm_bf16_wmma<64, 64, false, false>
            <<<(N + 127) / 128, 256, 0, stream>>>(h, conv_W + (size_t)l * H * H, nullptr, hw, N);
        // acc = conv_b[l] broadcast
        init_acc<<<nhBlocks, 256, 0, stream>>>(acc, conv_b + l * H, (int)NH);
        // acc[dst] += hw[src] * norm
        scatter_kernel<<<scBlocks, 256, 0, stream>>>(hw, src, dst, norm, acc, E);
        // residual + BN(eval) + relu
        post_kernel<<<nhBlocks, 256, 0, stream>>>(acc, hlast, h,
            bn_g + l * H, bn_b + l * H, bn_m + l * H, bn_v + l * H, (int)NH, l > 0 ? 1 : 0);
    }

    // --- predictor: out = h @ pred_W + pred_b  (C=40, padded to 48 in LDS) ---
    gemm_bf16_wmma<64, 40, true, false>
        <<<(N + 127) / 128, 256, 0, stream>>>(h, pred_W, pred_b, out, N);
}
